// SelfAttention_3951369912518
// MI455X (gfx1250) — compile-verified, matrix operand order checked by reading, not run on hardware
//
#include <hip/hip_runtime.h>

// ---------------------------------------------------------------------------
// MHA forward for MI455X (gfx1250, wave32, WMMA).
// B=4, T=2048, H=8, head_dim = EMB = 128.
// Flash-attention style: scores never hit memory. All matmuls use
// v_wmma_f32_16x16x32_bf16 with f32 accumulation.  K/V tiles are staged to
// LDS with double-buffered GLOBAL_LOAD_ASYNC_TO_LDS_B128 (ASYNCcnt).
// ---------------------------------------------------------------------------

typedef __bf16 bf16;
typedef __attribute__((ext_vector_type(16))) __bf16 v16bf;
typedef __attribute__((ext_vector_type(8)))  __bf16 v8bf;
typedef __attribute__((ext_vector_type(8)))  float  v8f;

#define BATCH 4
#define SEQ   2048
#define HEADS 8
#define EMB   128
#define NBH   (BATCH * HEADS)          // 32
#define PROJ  (HEADS * EMB)            // 1024

// scale folded into Wq: (1/sqrt(EMB)) for q*k, times log2(e) so softmax can
// use exp2 (v_exp_f32) directly.  1.4426950408889634 / sqrt(128)
#define QSCALE 0.12751724f

// LDS staging for the attention kernel (per 32-key iteration, double buffered)
#define KROWB 272                       // 256B K row + 16B pad (bank+4/row)
#define VROWB 80                        // 64B  V row + 16B pad (bank+20/row)
#define KBUFB (32 * KROWB)              // 8704 B
#define VBUFB (128 * VROWB)             // 10240 B
#define BUFB  (KBUFB + VBUFB)           // 18944 B per buffer

// D = A(16x32 bf16) * B(32x16 bf16) + C(16x16 f32)
__device__ __forceinline__ v8f wmma_bf16(v16bf a, v16bf b, v8f c) {
  return __builtin_amdgcn_wmma_f32_16x16x32_bf16(false, a, false, b,
                                                 (short)0, c, false, false);
}

// Per-lane A/B operand fragment for one 32-wide k-chunk.
// ISA layout (16-bit A 16x32): lanes 0-15 hold k = [0..7],[16..23];
// lanes 16-31 hold k = [8..15],[24..31].  => two contiguous 8xbf16 runs.
__device__ __forceinline__ v16bf ld_tile_bf16(const bf16* p, int khalf) {
  v8bf lo = *(const v8bf*)(p + khalf * 8);
  v8bf hi = *(const v8bf*)(p + 16 + khalf * 8);
  v16bf r;
#pragma unroll
  for (int i = 0; i < 8; ++i) { r[i] = lo[i]; r[i + 8] = hi[i]; }
  return r;
}

__device__ __forceinline__ v16bf ld_tile_f32(const float* p, int khalf) {
  v16bf r;
#pragma unroll
  for (int i = 0; i < 8; ++i) {
    r[i]     = (bf16)p[khalf * 8 + i];
    r[i + 8] = (bf16)p[16 + khalf * 8 + i];
  }
  return r;
}

__device__ __forceinline__ void async_b128(unsigned lds_off, const void* gptr) {
  asm volatile("global_load_async_to_lds_b128 %0, %1, off"
               :: "v"(lds_off), "v"((unsigned long long)(uintptr_t)gptr)
               : "memory");
}

// ---------------------------------------------------------------------------
// Kernel 0: transpose + convert weights to bf16.
//   Wq,Wk,Wv (128x1024 f32) -> Wt[mat] (1024x128 bf16)   (Wq scaled by QSCALE)
//   Wu       (1024x128 f32) -> Wut     (128x1024 bf16)
// ---------------------------------------------------------------------------
__global__ void __launch_bounds__(256) prep_weights_kernel(
    const float* __restrict__ Wq, const float* __restrict__ Wk,
    const float* __restrict__ Wv, const float* __restrict__ Wu,
    bf16* __restrict__ Wt, bf16* __restrict__ Wut) {
  int id = blockIdx.x * 256 + threadIdx.x;          // 0 .. 4*131072-1
  int mat = id >> 17;
  int r   = id & 131071;
  if (mat < 3) {
    const float* W = (mat == 0) ? Wq : ((mat == 1) ? Wk : Wv);
    int n = r >> 7;          // out column in [0,1024)
    int k = r & 127;         // in  row    in [0,128)
    float v = W[(size_t)k * PROJ + n];
    if (mat == 0) v *= QSCALE;
    Wt[(size_t)mat * 131072 + r] = (bf16)v;
  } else {
    int n = r >> 10;         // out column in [0,128)
    int k = r & 1023;        // in  row    in [0,1024)
    Wut[r] = (bf16)Wu[(size_t)k * EMB + n];
  }
}

// ---------------------------------------------------------------------------
// Kernel 1: QKV projection.  Y = x(8192x128) @ W(128x1024), bf16 WMMA.
// Q,K -> [bh][T][128] bf16.   V -> transposed Vt [bh][128][T] bf16.
// grid (512 row-tiles, 16, 3 mats), block 128 (4 waves, each one col-tile).
// ---------------------------------------------------------------------------
__global__ void __launch_bounds__(128) qkv_proj_kernel(
    const float* __restrict__ x, const bf16* __restrict__ Wt,
    bf16* __restrict__ Qb, bf16* __restrict__ Kb, bf16* __restrict__ Vt) {
  const int lane  = threadIdx.x & 31;
  const int wave  = threadIdx.x >> 5;
  const int khalf = lane >> 4;
  const int l16   = lane & 15;
  const int rt    = blockIdx.x;                 // row tile: 16 tokens
  const int ct    = blockIdx.y * 4 + wave;      // col tile: 16 of 1024
  const int mat   = blockIdx.z;                 // 0=Q 1=K 2=V

  const float* xp = x + (size_t)(rt * 16 + l16) * EMB;                 // A row
  const bf16*  wp = Wt + (size_t)mat * 131072 + (size_t)(ct * 16 + l16) * EMB;

  v8f acc = {};
#pragma unroll
  for (int c = 0; c < 4; ++c)
    acc = wmma_bf16(ld_tile_f32(xp + c * 32, khalf),
                    ld_tile_bf16(wp + c * 32, khalf), acc);

  const int cfull = ct * 16 + l16;
  const int h = cfull >> 7;
  const int d = cfull & 127;
  const int tflat0 = rt * 16 + 8 * khalf;       // token for v=0
  const int b  = tflat0 >> 11;                  // constant across v (T%16==0)
  const int t0 = tflat0 & 2047;
  const size_t bh = (size_t)(b * HEADS + h);

  if (mat == 2) {
    v8bf ov;
#pragma unroll
    for (int v = 0; v < 8; ++v) ov[v] = (bf16)acc[v];
    *(v8bf*)&Vt[(bh * EMB + d) * SEQ + t0] = ov;   // 8 consecutive tokens
  } else {
    bf16* dst = (mat == 0) ? Qb : Kb;
#pragma unroll
    for (int v = 0; v < 8; ++v)
      dst[(bh * SEQ + (t0 + v)) * EMB + d] = (bf16)acc[v];
  }
}

// ---------------------------------------------------------------------------
// Kernel 2: fused flash attention.  One wave = 16 queries, full d=128.
// S^T tiles = K(16x32) x Q^T(32x16): query index lands in the lane dim, so
// softmax reductions are in-lane + one shfl_xor(16).  P^T C-tiles map
// directly onto the B-operand packing of O^T = V^T x P^T (pure in-lane).
// The block's 4 waves share each 32-key K/V tile via double-buffered
// async global->LDS copies (ASYNCcnt), so L2 traffic drops 4x and the next
// tile's fill overlaps this tile's WMMAs.
// grid (T/64 = 32, NBH = 32), block 128.
// ---------------------------------------------------------------------------
__global__ void __launch_bounds__(128) attn_kernel(
    const bf16* __restrict__ Qb, const bf16* __restrict__ Kb,
    const bf16* __restrict__ Vt, bf16* __restrict__ O) {
  __shared__ char smem[2 * BUFB];                 // ~37 KB of the 320 KB WGP LDS

  const int tid   = threadIdx.x;
  const int lane  = tid & 31;
  const int wave  = tid >> 5;
  const int khalf = lane >> 4;
  const int l16   = lane & 15;
  const int qtile = blockIdx.x * 4 + wave;      // 0..127
  const int bh    = blockIdx.y;                 // 0..31
  const int b     = bh >> 3;
  const int h     = bh & 7;

  const char* kgl0 = (const char*)(Kb + (size_t)bh * SEQ * EMB);   // K tile rows contiguous
  const char* vgl0 = (const char*)(Vt + (size_t)bh * EMB * SEQ);   // V rows stride SEQ*2
  const unsigned ldsBase = (unsigned)(uintptr_t)smem;

  // ---- async fill of one 32-key tile (K: 8 KB contiguous, V: 128 x 64 B) ---
  auto issue_tile = [&](int j, int bufsel) {
    const unsigned lb = ldsBase + bufsel * BUFB;
    const char* kg = kgl0 + (size_t)j * 32 * EMB * 2;
    const char* vg = vgl0 + (size_t)j * 32 * 2;
#pragma unroll
    for (int i = 0; i < 4; ++i) {                 // K: 512 16B chunks / block
      int c = i * 128 + tid;
      int row = c >> 4, col = c & 15;
      async_b128(lb + row * KROWB + col * 16, kg + row * 256 + col * 16);
    }
#pragma unroll
    for (int i = 0; i < 4; ++i) {                 // V: 512 16B chunks / block
      int c = i * 128 + tid;
      int row = c >> 2, q4 = c & 3;
      async_b128(lb + KBUFB + row * VROWB + q4 * 16,
                 vg + (size_t)row * (SEQ * 2) + q4 * 16);
    }
  };

  // Q as B-operand (lane = query column), resident for the whole kernel.
  const bf16* qp = Qb + ((size_t)bh * SEQ + qtile * 16 + l16) * EMB;
  v16bf qreg[4];
#pragma unroll
  for (int c = 0; c < 4; ++c) qreg[c] = ld_tile_bf16(qp + c * 32, khalf);

  v8f o[8];
#pragma unroll
  for (int dt = 0; dt < 8; ++dt) o[dt] = (v8f){};
  float mrun = -1e30f, lrun = 0.f;

  const int NJ = SEQ / 32;
  issue_tile(0, 0);                               // prologue fill

  for (int j = 0; j < NJ; ++j) {
    // overlap: start next tile's fill, then drain this tile's 8 async ops
    if (j + 1 < NJ) {
      issue_tile(j + 1, (j + 1) & 1);
      asm volatile("s_wait_asynccnt 0x8" ::: "memory");
    } else {
      asm volatile("s_wait_asynccnt 0x0" ::: "memory");
    }
    __syncthreads();                              // other waves' fills done too

    const char* kb = smem + (j & 1) * BUFB;
    const char* vb = kb + KBUFB;

    v8f s0 = {}, s1 = {};
#pragma unroll
    for (int c = 0; c < 4; ++c)
      s0 = wmma_bf16(ld_tile_bf16((const bf16*)(kb + (l16)      * KROWB + c * 64), khalf),
                     qreg[c], s0);
#pragma unroll
    for (int c = 0; c < 4; ++c)
      s1 = wmma_bf16(ld_tile_bf16((const bf16*)(kb + (16 + l16) * KROWB + c * 64), khalf),
                     qreg[c], s1);

    // --- online softmax over this 32-key chunk (base-2; scale folded in Wq)
    float lm = s0[0];
#pragma unroll
    for (int v = 1; v < 8; ++v) lm = fmaxf(lm, s0[v]);
#pragma unroll
    for (int v = 0; v < 8; ++v) lm = fmaxf(lm, s1[v]);
    lm = fmaxf(lm, __shfl_xor(lm, 16, 32));      // combine key halves
    const float mnew = fmaxf(mrun, lm);
    const float corr = __builtin_exp2f(mrun - mnew);

    float p0[8], p1[8], ls = 0.f;
#pragma unroll
    for (int v = 0; v < 8; ++v) { p0[v] = __builtin_exp2f(s0[v] - mnew); ls += p0[v]; }
#pragma unroll
    for (int v = 0; v < 8; ++v) { p1[v] = __builtin_exp2f(s1[v] - mnew); ls += p1[v]; }
    ls += __shfl_xor(ls, 16, 32);
    lrun = lrun * corr + ls;
    mrun = mnew;

    // rescale running O^T accumulators (per-lane factor: lane = query)
#pragma unroll
    for (int dt = 0; dt < 8; ++dt)
#pragma unroll
      for (int v = 0; v < 8; ++v) o[dt][v] *= corr;

    // pack P^T C-tiles into the PV B-operand (pure in-lane, by ISA layout)
    v16bf pb;
#pragma unroll
    for (int e = 0; e < 8; ++e) { pb[e] = (bf16)p0[e]; pb[e + 8] = (bf16)p1[e]; }

    // O^T += V^T x P^T   (8 d-tiles of 16)
#pragma unroll
    for (int dt = 0; dt < 8; ++dt)
      o[dt] = wmma_bf16(ld_tile_bf16((const bf16*)(vb + (dt * 16 + l16) * VROWB), khalf),
                        pb, o[dt]);

    __syncthreads();                              // done reading this buffer
  }

  // normalize and store O[token][h*128 + d] (8 consecutive d per store)
  const float inv = 1.0f / lrun;
  bf16* orow = O + (size_t)(b * SEQ + qtile * 16 + l16) * PROJ + h * EMB;
#pragma unroll
  for (int dt = 0; dt < 8; ++dt) {
    v8bf ov;
#pragma unroll
    for (int v = 0; v < 8; ++v) ov[v] = (bf16)(o[dt][v] * inv);
    *(v8bf*)(orow + dt * 16 + khalf * 8) = ov;
  }
}

// ---------------------------------------------------------------------------
// Kernel 3: output projection. out(8192x128 f32) = O(8192x1024) @ Wu + bu.
// grid (512 row-tiles, 2), block 128 (4 waves, each one 16-col tile).
// ---------------------------------------------------------------------------
__global__ void __launch_bounds__(128) out_proj_kernel(
    const bf16* __restrict__ O, const bf16* __restrict__ Wut,
    const float* __restrict__ bu, float* __restrict__ out) {
  const int lane  = threadIdx.x & 31;
  const int wave  = threadIdx.x >> 5;
  const int khalf = lane >> 4;
  const int l16   = lane & 15;
  const int rt    = blockIdx.x;                 // 16 tokens
  const int ct    = blockIdx.y * 4 + wave;      // 16 of 128 out cols

  const bf16* op = O   + (size_t)(rt * 16 + l16) * PROJ;
  const bf16* wp = Wut + (size_t)(ct * 16 + l16) * PROJ;

  v8f acc = {};
#pragma unroll
  for (int c = 0; c < 32; ++c)
    acc = wmma_bf16(ld_tile_bf16(op + c * 32, khalf),
                    ld_tile_bf16(wp + c * 32, khalf), acc);

  const int col  = ct * 16 + l16;
  const float bc = bu[col];
  const int row0 = rt * 16 + 8 * khalf;
#pragma unroll
  for (int v = 0; v < 8; ++v)
    out[(size_t)(row0 + v) * EMB + col] = acc[v] + bc;
}

// ---------------------------------------------------------------------------
extern "C" void kernel_launch(void* const* d_in, const int* in_sizes, int n_in,
                              void* d_out, int out_size, void* d_ws, size_t ws_size,
                              hipStream_t stream) {
  const float* x  = (const float*)d_in[0];
  const float* Wq = (const float*)d_in[1];
  const float* Wk = (const float*)d_in[2];
  const float* Wv = (const float*)d_in[3];
  const float* Wu = (const float*)d_in[4];
  const float* bu = (const float*)d_in[5];
  float* out = (float*)d_out;

  // workspace carve-out (byte offsets; 65 MB total)
  char* ws = (char*)d_ws;
  const size_t SZ_WT  = (size_t)3 * 1024 * 128 * sizeof(bf16);   // 768 KB
  const size_t SZ_WUT = (size_t)128 * 1024 * sizeof(bf16);       // 256 KB
  const size_t SZ_M   = (size_t)NBH * SEQ * EMB * sizeof(bf16);  // 16 MB each
  bf16* Wt  = (bf16*)(ws);
  bf16* Wut = (bf16*)(ws + SZ_WT);
  bf16* Qb  = (bf16*)(ws + SZ_WT + SZ_WUT);
  bf16* Kb  = (bf16*)((char*)Qb + SZ_M);
  bf16* Vt  = (bf16*)((char*)Kb + SZ_M);
  bf16* O   = (bf16*)((char*)Vt + SZ_M);

  prep_weights_kernel<<<2048, 256, 0, stream>>>(Wq, Wk, Wv, Wu, Wt, Wut);
  qkv_proj_kernel<<<dim3(BATCH * SEQ / 16, 16, 3), 128, 0, stream>>>(x, Wt, Qb, Kb, Vt);
  attn_kernel<<<dim3(SEQ / 64, NBH), 128, 0, stream>>>(Qb, Kb, Vt, O);
  out_proj_kernel<<<dim3(BATCH * SEQ / 16, 2), 128, 0, stream>>>(O, Wut, bu, out);
}
